// TriplaneImplicitGenerator3d_48911087566937
// MI455X (gfx1250) — matrix operand order repaired and checked by reading, not compiled, not run on hardware
//
#include <hip/hip_runtime.h>

// ---------------------------------------------------------------------------
// Triplane NeRF renderer for MI455X (gfx1250, wave32, WMMA).
// Decoder GEMM [P,32]x[32,52] runs on v_wmma_f32_16x16x32_f16.
// ---------------------------------------------------------------------------

#define IMG      128
#define NRAY     (IMG * IMG)       // 16384 per batch
#define BATCH    2
#define NSTEP    24
#define FEAT_C   32
#define OUT_C    52
#define RES      256
#define PLANE_SZ (RES * RES)       // 65536
#define PTOTAL   (BATCH * NRAY * NSTEP)   // 786432 samples per pass
#define NRAYS_T  (BATCH * NRAY)           // 32768 rays

typedef _Float16 v16h __attribute__((ext_vector_type(16)));
typedef _Float16 v8h  __attribute__((ext_vector_type(8)));
typedef float    v8f  __attribute__((ext_vector_type(8)));

// ---------------------------------------------------------------------------
// Kernel 0: pack W_dec [32,52] into CDNA5 WMMA B-fragment layout.
// B is 32x16 f16 per tile (4 tiles cover cols 0..63; cols >= 52 are zero).
// Layout per tile: lane 0-15 -> N=lane, K {0-7,16-23};
//                  lane 16-31 -> N=lane-16, K {8-15,24-31}.
// packB: [4 tiles][32 lanes][16 halfs]
// ---------------------------------------------------------------------------
__global__ void pack_wdec(const float* __restrict__ W, _Float16* __restrict__ packB) {
    int t = threadIdx.x;
    if (t >= 4 * 32) return;
    int tile = t >> 5;
    int lane = t & 31;
    int ncol = tile * 16 + (lane & 15);
    int kb   = (lane < 16) ? 0 : 8;
#pragma unroll
    for (int h = 0; h < 16; ++h) {
        float v = (ncol < OUT_C) ? W[(kb + h + ((h >> 3) << 3)) * OUT_C + ncol] : 0.0f;
        packB[((tile * 32 + lane) << 4) + h] = (_Float16)v;
    }
}

// ---------------------------------------------------------------------------
// Bilinear border-mode sample of one 32-channel plane block; accumulate.
// ---------------------------------------------------------------------------
__device__ __forceinline__ void sample_plane(const float* __restrict__ p,
                                             float cx, float cy,
                                             float* __restrict__ acc) {
    float gx = (cx + 1.0f) * 0.5f * (float)(RES - 1);
    float gy = (cy + 1.0f) * 0.5f * (float)(RES - 1);
    gx = fminf(fmaxf(gx, 0.0f), (float)(RES - 1));
    gy = fminf(fmaxf(gy, 0.0f), (float)(RES - 1));
    int x0 = (int)floorf(gx);
    int y0 = (int)floorf(gy);
    int x1 = min(x0 + 1, RES - 1);
    int y1 = min(y0 + 1, RES - 1);
    float wx = gx - (float)x0;
    float wy = gy - (float)y0;
    float w00 = (1.0f - wx) * (1.0f - wy);
    float w01 = wx * (1.0f - wy);
    float w10 = (1.0f - wx) * wy;
    float w11 = wx * wy;
    int i00 = y0 * RES + x0;
    int i01 = y0 * RES + x1;
    int i10 = y1 * RES + x0;
    int i11 = y1 * RES + x1;
#pragma unroll
    for (int c = 0; c < FEAT_C; ++c) {
        const float* pc = p + (long)c * PLANE_SZ;
        acc[c] = fmaf(pc[i00], w00,
                 fmaf(pc[i01], w01,
                 fmaf(pc[i10], w10,
                 fmaf(pc[i11], w11, acc[c]))));
    }
}

// ---------------------------------------------------------------------------
// Kernel 1/4: per-sample triplane gather -> f16 feature rows [P, 32].
// zbuf has layout [B*N*S] (works for both coarse z_vals and fine_z).
// ---------------------------------------------------------------------------
__global__ void gather_feat(const float* __restrict__ triplane,
                            const float* __restrict__ ray_o,
                            const float* __restrict__ ray_d,
                            const float* __restrict__ zbuf,
                            _Float16* __restrict__ featOut) {
    int idx = blockIdx.x * blockDim.x + threadIdx.x;
    if (idx >= PTOTAL) return;
    int ray = idx / NSTEP;
    int b   = ray / NRAY;

    float z  = zbuf[idx];
    float px = fmaf(ray_d[ray * 3 + 0], z, ray_o[ray * 3 + 0]);
    float py = fmaf(ray_d[ray * 3 + 1], z, ray_o[ray * 3 + 1]);
    float pz = fmaf(ray_d[ray * 3 + 2], z, ray_o[ray * 3 + 2]);

    const float inv = 1.0f / (0.185f + 1e-5f);
    const float lo = -1.0f + 1e-5f, hi = 1.0f - 1e-5f;
    float cx = fminf(fmaxf(px * inv, lo), hi);
    float cy = fminf(fmaxf(py * inv, lo), hi);
    float cz = fminf(fmaxf(pz * inv, lo), hi);

    float acc[FEAT_C];
#pragma unroll
    for (int c = 0; c < FEAT_C; ++c) acc[c] = 0.0f;

    const float* base = triplane + (long)b * 3 * FEAT_C * PLANE_SZ;
    sample_plane(base + 0L * FEAT_C * PLANE_SZ, cx, cz, acc);   // xz plane
    sample_plane(base + 1L * FEAT_C * PLANE_SZ, cx, cy, acc);   // xy plane
    sample_plane(base + 2L * FEAT_C * PLANE_SZ, cy, cz, acc);   // yz plane

    v8h h0, h1, h2, h3;
#pragma unroll
    for (int i = 0; i < 8; ++i) {
        h0[i] = (_Float16)acc[i];
        h1[i] = (_Float16)acc[i + 8];
        h2[i] = (_Float16)acc[i + 16];
        h3[i] = (_Float16)acc[i + 24];
    }
    v8h* dst = (v8h*)(featOut + (long)idx * FEAT_C);
    dst[0] = h0; dst[1] = h1; dst[2] = h2; dst[3] = h3;
}

// ---------------------------------------------------------------------------
// Kernel 2/5: WMMA decoder. One wave = 16 samples x 64 cols (52 valid).
// A fragment (16x32 f16, ISA layout): lane 0-15 -> row M=lane, K {0-7,16-23};
//                                     lane 16-31 -> row M=lane-16, K {8-15,24-31}.
// C/D fragment: VGPR v -> row M=v (lanes 0-15) or v+8 (lanes 16-31), N=lane%16.
// ---------------------------------------------------------------------------
__global__ void decode_wmma(const _Float16* __restrict__ feat,
                            const _Float16* __restrict__ packB,
                            const float* __restrict__ b_dec,
                            float* __restrict__ out) {
    int gtid = blockIdx.x * blockDim.x + threadIdx.x;
    int wave = gtid >> 5;
    int lane = threadIdx.x & 31;
    long base = (long)wave * 16;
    if (base >= PTOTAL) return;

    // ---- load A fragment: two 16-byte chunks from this lane's sample row
    int  row  = (int)base + (lane & 15);
    int  koff = (lane < 16) ? 0 : 8;
    const _Float16* fr = feat + (long)row * FEAT_C;
    v8h alo = *(const v8h*)(fr + koff);        // K = koff .. koff+7
    v8h ahi = *(const v8h*)(fr + koff + 16);   // K = koff+16 .. koff+23
    v16h a;
#pragma unroll
    for (int i = 0; i < 8; ++i) { a[i] = alo[i]; a[i + 8] = ahi[i]; }

    int colLane = lane & 15;
#pragma unroll
    for (int t = 0; t < 4; ++t) {
        v16h bm = *(const v16h*)(packB + (((t << 5) + lane) << 4));
        v8f c = {};
        c = __builtin_amdgcn_wmma_f32_16x16x32_f16(
                /*neg_a=*/false, a, /*neg_b=*/false, bm,
                /*c_mod=*/(short)0, c, /*reuse_a=*/false, /*reuse_b=*/false);
        int col = t * 16 + colLane;
        if (col < OUT_C) {
            float bias = b_dec[col];
#pragma unroll
            for (int v = 0; v < 8; ++v) {
                int m = (lane < 16) ? v : (v + 8);
                out[(base + m) * OUT_C + col] = c[v] + bias;
            }
        }
    }
}

// ---------------------------------------------------------------------------
// Kernel 3: per-ray coarse integration weights + inverse-CDF fine sampling.
// One lane per ray.
// ---------------------------------------------------------------------------
__global__ void pdf_kernel(const float* __restrict__ z_vals,
                           const float* __restrict__ coarse_out,
                           float* __restrict__ fine_z) {
    int r = blockIdx.x * blockDim.x + threadIdx.x;
    if (r >= NRAYS_T) return;

    float z[NSTEP];
#pragma unroll
    for (int s = 0; s < NSTEP; ++s) z[s] = z_vals[(long)r * NSTEP + s];

    float w[NSTEP];
    float T = 1.0f;
#pragma unroll
    for (int s = 0; s < NSTEP; ++s) {
        float delta = (s < NSTEP - 1) ? (z[s + 1] - z[s]) : 1e10f;
        float sg = fmaxf(coarse_out[((long)r * NSTEP + s) * OUT_C + FEAT_C], 0.0f);
        float alpha = 1.0f - __expf(-delta * sg);
        w[s] = alpha * T + 1e-5f;
        T *= (1.0f - alpha + 1e-10f);
    }

    // pdf over w[1..22] (22 bins), cdf[0..22]
    float sum = 0.0f;
#pragma unroll
    for (int i = 1; i <= NSTEP - 2; ++i) sum += w[i];
    float rsum = 1.0f / sum;

    float cdf[NSTEP - 1];   // 23 entries
    cdf[0] = 0.0f;
#pragma unroll
    for (int i = 1; i <= NSTEP - 2; ++i) cdf[i] = cdf[i - 1] + w[i] * rsum;

    const float du = 1.0f / (float)(NSTEP - 1);
#pragma unroll 1
    for (int j = 0; j < NSTEP; ++j) {
        float u = (float)j * du;
        int ind = NSTEP - 1;                 // searchsorted(cdf, u, 'right')
        for (int i = 0; i <= NSTEP - 2; ++i) {
            if (cdf[i] > u) { ind = i; break; }
        }
        int below = min(max(ind - 1, 0), NSTEP - 2);
        int above = min(ind, NSTEP - 2);
        float cb = cdf[below], ca = cdf[above];
        float d  = ca - cb;
        float denom = (d < 1e-5f) ? 1.0f : d;
        float t = (u - cb) / denom;
        float bb = 0.5f * (z[below] + z[below + 1]);
        float ba = 0.5f * (z[above] + z[above + 1]);
        fine_z[(long)r * NSTEP + j] = bb + t * (ba - bb);
    }
}

// ---------------------------------------------------------------------------
// Kernel 6: per-ray merge of sorted coarse+fine depths (stable: fine wins
// ties, matching argsort over concat([fine, coarse])) + final integration.
// Writes [B, 51, 128, 128] (channels: 32 feat + 19 label), coalesced per ch.
// ---------------------------------------------------------------------------
__global__ void integrate_final(const float* __restrict__ z_vals,
                                const float* __restrict__ fine_z,
                                const float* __restrict__ coarse_out,
                                const float* __restrict__ fine_out,
                                float* __restrict__ out) {
    int r = blockIdx.x * blockDim.x + threadIdx.x;
    if (r >= NRAYS_T) return;
    int b = r / NRAY;
    int n = r % NRAY;

    float featAcc[FEAT_C];
    float labAcc[19];
#pragma unroll
    for (int c = 0; c < FEAT_C; ++c) featAcc[c] = 0.0f;
#pragma unroll
    for (int c = 0; c < 19; ++c) labAcc[c] = 0.0f;

    const long rb = (long)r * NSTEP;
    int fi = 0, ci = 0;
    float headF = fine_z[rb];
    float headC = z_vals[rb];
    float T = 1.0f;

#pragma unroll 1
    for (int k = 0; k < 2 * NSTEP; ++k) {
        bool takeFine = (fi < NSTEP) && (ci >= NSTEP || headF <= headC);
        float curz;
        const float* src;
        long rowoff;
        if (takeFine) {
            curz = headF;
            src = fine_out;
            rowoff = (rb + fi) * OUT_C;
            ++fi;
            headF = (fi < NSTEP) ? fine_z[rb + fi] : 3.4e38f;
        } else {
            curz = headC;
            src = coarse_out;
            rowoff = (rb + ci) * OUT_C;
            ++ci;
            headC = (ci < NSTEP) ? z_vals[rb + ci] : 3.4e38f;
        }
        float nextz = fminf(headF, headC);
        float delta = (k < 2 * NSTEP - 1) ? (nextz - curz) : 1e10f;
        float sg = fmaxf(src[rowoff + FEAT_C], 0.0f);
        float alpha = 1.0f - __expf(-delta * sg);
        float wgt = alpha * T;
        T *= (1.0f - alpha + 1e-10f);
#pragma unroll
        for (int c = 0; c < FEAT_C; ++c) featAcc[c] = fmaf(wgt, src[rowoff + c], featAcc[c]);
#pragma unroll
        for (int c = 0; c < 19; ++c) labAcc[c] = fmaf(wgt, src[rowoff + FEAT_C + 1 + c], labAcc[c]);
    }

    long ob = (long)b * 51 * NRAY + n;
#pragma unroll
    for (int c = 0; c < FEAT_C; ++c) out[ob + (long)c * NRAY] = featAcc[c];
#pragma unroll
    for (int c = 0; c < 19; ++c) out[ob + (long)(FEAT_C + c) * NRAY] = labAcc[c];
}

// ---------------------------------------------------------------------------
// Launcher. Workspace layout (bytes):
//   [0, 4K)                          packB      (f16, 4x32x16)
//   [4K, 4K+50.3M)                   featbuf    (f16, P*32)   -- reused by both passes
//   next 163.6M                      coarse_out (f32, P*52)
//   next 163.6M                      fine_out   (f32, P*52)
//   next 3.1M                        fine_z     (f32, P)
// ---------------------------------------------------------------------------
extern "C" void kernel_launch(void* const* d_in, const int* in_sizes, int n_in,
                              void* d_out, int out_size, void* d_ws, size_t ws_size,
                              hipStream_t stream) {
    const float* triplane = (const float*)d_in[0];
    const float* W_dec    = (const float*)d_in[1];
    const float* b_dec    = (const float*)d_in[2];
    const float* ray_o    = (const float*)d_in[3];
    const float* ray_d    = (const float*)d_in[4];
    const float* z_vals   = (const float*)d_in[5];
    float* out = (float*)d_out;

    char* ws = (char*)d_ws;
    size_t off = 0;
    _Float16* packB = (_Float16*)(ws + off);           off += 4096;
    _Float16* featbuf = (_Float16*)(ws + off);         off += (size_t)PTOTAL * FEAT_C * sizeof(_Float16);
    float* coarse_out = (float*)(ws + off);            off += (size_t)PTOTAL * OUT_C * sizeof(float);
    float* fine_out = (float*)(ws + off);              off += (size_t)PTOTAL * OUT_C * sizeof(float);
    float* fine_z = (float*)(ws + off);                off += (size_t)PTOTAL * sizeof(float);
    (void)off; (void)ws_size; (void)in_sizes; (void)n_in; (void)out_size;

    // K0: pack decoder weights into WMMA B-fragment layout
    hipLaunchKernelGGL(pack_wdec, dim3(1), dim3(128), 0, stream, W_dec, packB);

    // K1: coarse triplane gather
    {
        int threads = 256;
        int blocks = (PTOTAL + threads - 1) / threads;
        hipLaunchKernelGGL(gather_feat, dim3(blocks), dim3(threads), 0, stream,
                           triplane, ray_o, ray_d, z_vals, featbuf);
    }
    // K2: coarse WMMA decode
    {
        int threads = 256;                       // 8 waves/block, 16 samples/wave
        long waves = PTOTAL / 16;
        int blocks = (int)((waves * 32 + threads - 1) / threads);
        hipLaunchKernelGGL(decode_wmma, dim3(blocks), dim3(threads), 0, stream,
                           featbuf, packB, b_dec, coarse_out);
    }
    // K3: per-ray weights + inverse-CDF fine depths
    {
        int threads = 256;
        int blocks = (NRAYS_T + threads - 1) / threads;
        hipLaunchKernelGGL(pdf_kernel, dim3(blocks), dim3(threads), 0, stream,
                           z_vals, coarse_out, fine_z);
    }
    // K4: fine triplane gather (reuses featbuf)
    {
        int threads = 256;
        int blocks = (PTOTAL + threads - 1) / threads;
        hipLaunchKernelGGL(gather_feat, dim3(blocks), dim3(threads), 0, stream,
                           triplane, ray_o, ray_d, fine_z, featbuf);
    }
    // K5: fine WMMA decode
    {
        int threads = 256;
        long waves = PTOTAL / 16;
        int blocks = (int)((waves * 32 + threads - 1) / threads);
        hipLaunchKernelGGL(decode_wmma, dim3(blocks), dim3(threads), 0, stream,
                           featbuf, packB, b_dec, fine_out);
    }
    // K6: merge + final integration + transposed output
    {
        int threads = 256;
        int blocks = (NRAYS_T + threads - 1) / threads;
        hipLaunchKernelGGL(integrate_final, dim3(blocks), dim3(threads), 0, stream,
                           z_vals, fine_z, coarse_out, fine_out, out);
    }
}
